// GNNModel_55619826483422
// MI455X (gfx1250) — matrix operand order repaired and checked by reading, compiled, not run on hardware
//
#include <hip/hip_runtime.h>
#include <hip/hip_bf16.h>

typedef _Float16 v16h __attribute__((ext_vector_type(16)));
typedef _Float16 v8h  __attribute__((ext_vector_type(8)));
typedef _Float16 v4h  __attribute__((ext_vector_type(4)));
typedef float    v8f  __attribute__((ext_vector_type(8)));

#define HDIM 128

// Guaranteed-native f32 scatter atomic (GLOBAL_ATOMIC_ADD_F32, no CAS loop).
__device__ __forceinline__ void atomicAddF32(float* p, float v) {
    asm volatile("global_atomic_add_f32 %0, %1, off"
                 :: "v"(p), "v"(v)
                 : "memory");
}

// ---------------- small utility kernels ----------------

__global__ void zero_f32(float* __restrict__ p, int n) {
    int i = blockIdx.x * blockDim.x + threadIdx.x;
    if (i < n) p[i] = 0.0f;
}

__global__ void fill_f32(float* __restrict__ p, float v, int n) {
    int i = blockIdx.x * blockDim.x + threadIdx.x;
    if (i < n) p[i] = v;
}

// vectorized f32 -> f16 (n must be a multiple of 4)
__global__ void cvt_f32_to_f16_v4(const float4* __restrict__ in, v4h* __restrict__ out, int n4) {
    int i = blockIdx.x * blockDim.x + threadIdx.x;
    if (i < n4) {
        float4 v = in[i];
        v4h h;
        h[0] = (_Float16)v.x; h[1] = (_Float16)v.y;
        h[2] = (_Float16)v.z; h[3] = (_Float16)v.w;
        out[i] = h;
    }
}

// vectorized relu + f32 -> f16
__global__ void relu_cvt_f16_v4(const float4* __restrict__ in, v4h* __restrict__ out, int n4) {
    int i = blockIdx.x * blockDim.x + threadIdx.x;
    if (i < n4) {
        float4 v = in[i];
        v4h h;
        h[0] = (_Float16)fmaxf(v.x, 0.0f); h[1] = (_Float16)fmaxf(v.y, 0.0f);
        h[2] = (_Float16)fmaxf(v.z, 0.0f); h[3] = (_Float16)fmaxf(v.w, 0.0f);
        out[i] = h;
    }
}

// deg[dst] += ew  (self-loop "1" already pre-filled)
__global__ void deg_edges(const int* __restrict__ dst, const float* __restrict__ ew,
                          float* __restrict__ deg, int E) {
    int e = blockIdx.x * blockDim.x + threadIdx.x;
    if (e < E) atomicAddF32(&deg[dst[e]], ew[e]);
}

__global__ void deg_rsqrt(float* __restrict__ d, int n) {
    int i = blockIdx.x * blockDim.x + threadIdx.x;
    if (i < n) {
        float v = d[i];
        d[i] = (v > 0.0f) ? rsqrtf(v) : 0.0f;
    }
}

// ---------------- WMMA GEMM: Out[M x 128] = A[M x 128] * W^T, W is [128 x 128]
// One wave computes one 16x16 tile; 8 waves per block cover all 8 column tiles.
__global__ void gemm128_wmma(const _Float16* __restrict__ A,
                             const _Float16* __restrict__ W,   // W[n][k], B = W^T
                             float* __restrict__ Out,
                             const float* __restrict__ bias,   // may be null
                             int relu, int M) {
    const int lane   = threadIdx.x & 31;
    const int ntile  = threadIdx.x >> 5;     // 0..7
    const int mtile  = blockIdx.x;
    const int halfid = lane >> 4;            // 0 or 1
    const int r      = lane & 15;

    int arow = mtile * 16 + r;
    if (arow >= M) arow = M - 1;             // keep EXEC full for WMMA; stores are guarded
    const _Float16* arowp = A + (size_t)arow * HDIM;
    const _Float16* browp = W + (size_t)(ntile * 16 + r) * HDIM;

    v8f c = {};
#pragma unroll
    for (int k0 = 0; k0 < HDIM; k0 += 32) {
        // A fragment: row m = r, K = h*16 .. h*16+15 contiguous (matches B)
        v16h a = *(const v16h*)(arowp + k0 + halfid * 16);
        // B fragment: col n = r, K = h*16 .. h*16+15 contiguous in weight row
        v16h b = *(const v16h*)(browp + k0 + halfid * 16);
        c = __builtin_amdgcn_wmma_f32_16x16x32_f16(false, a, false, b, (short)0, c, false, false);
    }

    const int col = ntile * 16 + r;
    const float bv = bias ? bias[col] : 0.0f;
#pragma unroll
    for (int i = 0; i < 8; ++i) {
        int orow = mtile * 16 + halfid * 8 + i;   // C/D layout: VGPR i -> M = i + 8*halfid
        if (orow < M) {
            float v = c[i] + bv;
            if (relu) v = fmaxf(v, 0.0f);
            Out[(size_t)orow * HDIM + col] = v;
        }
    }
}

// out[i,:] = dinv[i]^2 * h[i,:] + bias[:]   (self-loop term + bias, before edge scatter)
// vectorized: one float4 per thread, total4 = N*128/4
__global__ void agg_init_v4(const float4* __restrict__ h, const float* __restrict__ dinv,
                            const float4* __restrict__ bias, float4* __restrict__ out, int total4) {
    int i = blockIdx.x * blockDim.x + threadIdx.x;
    if (i < total4) {
        int row = i >> 5;                 // 32 float4 per row of 128
        float di = dinv[row];
        float s = di * di;
        float4 v = h[i];
        float4 b = bias[i & 31];
        float4 o;
        o.x = s * v.x + b.x; o.y = s * v.y + b.y;
        o.z = s * v.z + b.z; o.w = s * v.w + b.w;
        out[i] = o;
    }
}

// one wave per edge: out[dst,:] += dinv[s]*ew*dinv[d] * h[src,:]
__global__ void agg_edges(const float* __restrict__ h, const int* __restrict__ src,
                          const int* __restrict__ dst, const float* __restrict__ ew,
                          const float* __restrict__ dinv, float* __restrict__ out, int E) {
    int e = blockIdx.x * (blockDim.x >> 5) + (threadIdx.x >> 5);
    if (e >= E) return;
    int lane = threadIdx.x & 31;
    int s = src[e], d = dst[e];
    float nrm = dinv[s] * ew[e] * dinv[d];
    float4 v = ((const float4*)(h + (size_t)s * HDIM))[lane];
    float* o = out + (size_t)d * HDIM + lane * 4;
    atomicAddF32(o + 0, nrm * v.x);
    atomicAddF32(o + 1, nrm * v.y);
    atomicAddF32(o + 2, nrm * v.z);
    atomicAddF32(o + 3, nrm * v.w);
}

// one wave per node: sums[batch,:] += relu(h[node,:]); counts[batch] += 1
__global__ void pool_acc(const float* __restrict__ h, const int* __restrict__ batch,
                         float* __restrict__ sums, float* __restrict__ counts, int N) {
    int node = blockIdx.x * (blockDim.x >> 5) + (threadIdx.x >> 5);
    if (node >= N) return;
    int lane = threadIdx.x & 31;
    int b = batch[node];
    float4 v = ((const float4*)(h + (size_t)node * HDIM))[lane];
    float* s = sums + (size_t)b * HDIM + lane * 4;
    atomicAddF32(s + 0, fmaxf(v.x, 0.0f));
    atomicAddF32(s + 1, fmaxf(v.y, 0.0f));
    atomicAddF32(s + 2, fmaxf(v.z, 0.0f));
    atomicAddF32(s + 3, fmaxf(v.w, 0.0f));
    if (lane == 0) atomicAddF32(counts + b, 1.0f);
}

__global__ void pool_finalize(const float* __restrict__ sums, const float* __restrict__ counts,
                              _Float16* __restrict__ out, int total) {
    int i = blockIdx.x * blockDim.x + threadIdx.x;
    if (i < total) {
        int g = i >> 7;
        out[i] = (_Float16)(sums[i] / fmaxf(counts[g], 1.0f));
    }
}

// one wave per graph: out[g] = dot(o[g,:], fW2[0,:]) + fb2
__global__ void final_dot(const float* __restrict__ o, const float* __restrict__ fW2,
                          const float* __restrict__ fb2, float* __restrict__ out, int G) {
    int g = blockIdx.x * (blockDim.x >> 5) + (threadIdx.x >> 5);
    if (g >= G) return;
    int lane = threadIdx.x & 31;
    float4 a = ((const float4*)(o + (size_t)g * HDIM))[lane];
    float4 w = ((const float4*)fW2)[lane];
    float p = a.x * w.x + a.y * w.y + a.z * w.z + a.w * w.w;
#pragma unroll
    for (int off = 16; off > 0; off >>= 1) p += __shfl_xor(p, off, 32);
    if (lane == 0) out[g] = p + fb2[0];
}

// ---------------- host-side orchestration ----------------

extern "C" void kernel_launch(void* const* d_in, const int* in_sizes, int n_in,
                              void* d_out, int out_size, void* d_ws, size_t ws_size,
                              hipStream_t stream) {
    const float* x    = (const float*)d_in[0];
    const int*   eidx = (const int*)d_in[1];
    const float* ew   = (const float*)d_in[2];
    const int*   batch= (const int*)d_in[3];
    const float* W1   = (const float*)d_in[4];
    const float* b1   = (const float*)d_in[5];
    const float* W2   = (const float*)d_in[6];
    const float* b2   = (const float*)d_in[7];
    const float* fW1  = (const float*)d_in[8];
    const float* fb1  = (const float*)d_in[9];
    const float* fW2  = (const float*)d_in[10];
    const float* fb2  = (const float*)d_in[11];
    float* out = (float*)d_out;

    const int N = in_sizes[3];          // nodes
    const int E = in_sizes[2];          // edges
    const int G = out_size;             // graphs
    const int* src = eidx;
    const int* dst = eidx + E;

    // workspace layout (256B aligned)
    char* ws = (char*)d_ws;
    size_t off = 0;
    auto alloc = [&](size_t bytes) { size_t o = off; off += (bytes + 255) & ~(size_t)255; return o; };
    float*    dinv    = (float*)   (ws + alloc((size_t)N * 4));
    float*    bufA    = (float*)   (ws + alloc((size_t)N * HDIM * 4));   // GEMM outputs
    float*    bufB    = (float*)   (ws + alloc((size_t)N * HDIM * 4));   // aggregation accum
    _Float16* hh      = (_Float16*)(ws + alloc((size_t)N * HDIM * 2));   // f16 features (x, then a1)
    _Float16* W1h     = (_Float16*)(ws + alloc((size_t)HDIM * HDIM * 2));
    _Float16* W2h     = (_Float16*)(ws + alloc((size_t)HDIM * HDIM * 2));
    _Float16* fW1h    = (_Float16*)(ws + alloc((size_t)HDIM * HDIM * 2));
    float*    sums    = (float*)   (ws + alloc((size_t)G * HDIM * 4));
    float*    counts  = (float*)   (ws + alloc((size_t)G * 4));
    _Float16* pooledh = (_Float16*)(ws + alloc((size_t)G * HDIM * 2));
    float*    obuf    = (float*)   (ws + alloc((size_t)G * HDIM * 4));
    (void)ws_size; (void)n_in;

    const int T = 256;
    const int NH = N * HDIM;
    const int GH = G * HDIM;
    const int WW = HDIM * HDIM;
    auto blocks = [](long n, int t) { return (unsigned)((n + t - 1) / t); };

    // zero pooling accumulators
    zero_f32<<<blocks(GH, T), T, 0, stream>>>(sums, GH);
    zero_f32<<<blocks(G, T), T, 0, stream>>>(counts, G);

    // degree -> dinv
    fill_f32<<<blocks(N, T), T, 0, stream>>>(dinv, 1.0f, N);           // self-loop weight
    deg_edges<<<blocks(E, T), T, 0, stream>>>(dst, ew, dinv, E);
    deg_rsqrt<<<blocks(N, T), T, 0, stream>>>(dinv, N);

    // f16 conversions (vectorized; all counts are multiples of 4)
    cvt_f32_to_f16_v4<<<blocks(NH / 4, T), T, 0, stream>>>((const float4*)x, (v4h*)hh, NH / 4);
    cvt_f32_to_f16_v4<<<blocks(WW / 4, T), T, 0, stream>>>((const float4*)W1, (v4h*)W1h, WW / 4);
    cvt_f32_to_f16_v4<<<blocks(WW / 4, T), T, 0, stream>>>((const float4*)W2, (v4h*)W2h, WW / 4);
    cvt_f32_to_f16_v4<<<blocks(WW / 4, T), T, 0, stream>>>((const float4*)fW1, (v4h*)fW1h, WW / 4);

    const unsigned mtilesN = (unsigned)((N + 15) / 16);
    const unsigned mtilesG = (unsigned)((G + 15) / 16);

    // ---- GCN layer 1 ----
    gemm128_wmma<<<mtilesN, T, 0, stream>>>(hh, W1h, bufA, nullptr, 0, N);
    agg_init_v4<<<blocks(NH / 4, T), T, 0, stream>>>((const float4*)bufA, dinv,
                                                     (const float4*)b1, (float4*)bufB, NH / 4);
    agg_edges<<<blocks((long)E * 32, T), T, 0, stream>>>(bufA, src, dst, ew, dinv, bufB, E);
    relu_cvt_f16_v4<<<blocks(NH / 4, T), T, 0, stream>>>((const float4*)bufB, (v4h*)hh, NH / 4);

    // ---- GCN layer 2 ----
    gemm128_wmma<<<mtilesN, T, 0, stream>>>(hh, W2h, bufA, nullptr, 0, N);
    agg_init_v4<<<blocks(NH / 4, T), T, 0, stream>>>((const float4*)bufA, dinv,
                                                     (const float4*)b2, (float4*)bufB, NH / 4);
    agg_edges<<<blocks((long)E * 32, T), T, 0, stream>>>(bufA, src, dst, ew, dinv, bufB, E);

    // ---- global mean pool (ReLU fused) ----
    pool_acc<<<blocks((long)N * 32, T), T, 0, stream>>>(bufB, batch, sums, counts, N);
    pool_finalize<<<blocks(GH, T), T, 0, stream>>>(sums, counts, pooledh, GH);

    // ---- MLP head ----
    gemm128_wmma<<<mtilesG, T, 0, stream>>>(pooledh, fW1h, obuf, fb1, 1, G);
    final_dot<<<blocks((long)G * 32, T), T, 0, stream>>>(obuf, fW2, fb2, out, G);
}
